// SkipGramModel_83322365542554
// MI455X (gfx1250) — matrix operand order repaired and checked by reading, not compiled, not run on hardware
//
#include <hip/hip_runtime.h>

// CDNA5 wave32 WMMA types
typedef __attribute__((ext_vector_type(16))) __bf16 v16bf;
typedef __attribute__((ext_vector_type(8)))  __bf16 bf16x8;
typedef __attribute__((ext_vector_type(4)))  __bf16 bf16x4;
typedef __attribute__((ext_vector_type(8)))  float  v8f;

#define EMB_DIM 128
#define NEG_K   10
#define WAVES_PER_BLOCK 8
#define TILE_M  16
#define BATCH   65536

union FragBF { v16bf v; bf16x8 h[2]; };

// x is pre-clipped to [-10,10]: exp(x) in [4.5e-5, 2.2e4], so the naive
// fast-math form is safe and costs ~4 VALU (v_exp_f32 / v_add / v_log_f32)
// instead of the ~35-op precise log1p expansion.
__device__ __forceinline__ float softplus_f(float x) {
    return __logf(1.0f + __expf(x));
}

// D-matrix diagonal: lane i (i<8) holds D[i][i] in acc[i];
// lane i+16 (i>=8) holds D[i][i] in acc[i-8].
__device__ __forceinline__ float diag_select(v8f acc, int reg) {
    float d = acc[0];
    d = (reg == 1) ? acc[1] : d;
    d = (reg == 2) ? acc[2] : d;
    d = (reg == 3) ? acc[3] : d;
    d = (reg == 4) ? acc[4] : d;
    d = (reg == 5) ? acc[5] : d;
    d = (reg == 6) ? acc[6] : d;
    d = (reg == 7) ? acc[7] : d;
    return d;
}

// Prefetch one 512B embedding row: 32 lanes x 16B spacing covers all four
// 128B cachelines with a single global_prefetch_b8 issue.
__device__ __forceinline__ void prefetch_row(const float* row, int lane) {
    __builtin_prefetch((const char*)row + lane * 16, 0, 3);
}

__global__ void __launch_bounds__(32 * WAVES_PER_BLOCK)
sgns_loss_kernel(const float* __restrict__ u_emb,
                 const float* __restrict__ v_emb,
                 const int*   __restrict__ pos_u,
                 const int*   __restrict__ pos_v,
                 const int*   __restrict__ neg_v,
                 float*       __restrict__ block_sums)
{
    // per-wave staging: U tile row-major, B tile stored [N][K], both bf16
    __shared__ __align__(16) __bf16 sU[WAVES_PER_BLOCK][TILE_M * EMB_DIM]; // 4 KB/wave
    __shared__ __align__(16) __bf16 sB[WAVES_PER_BLOCK][EMB_DIM * TILE_M]; // 4 KB/wave
    __shared__ float s_wavesum[WAVES_PER_BLOCK];

    const int lane = threadIdx.x & 31;
    const int w    = threadIdx.x >> 5;
    const int tile = blockIdx.x * WAVES_PER_BLOCK + w;
    const int b0   = tile * TILE_M;

    __bf16* su = sU[w];
    __bf16* sb = sB[w];

    // ---- Prefetch the positive-v tile rows (consumed by mm == 0) ----
    #pragma unroll
    for (int i = 0; i < TILE_M; ++i) {
        const int idx = pos_v[b0 + i];                         // wave-uniform
        prefetch_row(v_emb + (size_t)idx * EMB_DIM, lane);
    }

    // ---- Gather U tile: 16 rows x 128 f32 (coalesced float4/lane) -> bf16 LDS ----
    #pragma unroll
    for (int i = 0; i < TILE_M; ++i) {
        const int idx = pos_u[b0 + i];                         // wave-uniform -> s_load
        const float4 f = *((const float4*)(u_emb + (size_t)idx * EMB_DIM) + lane);
        bf16x4 hh;
        hh[0] = (__bf16)f.x; hh[1] = (__bf16)f.y;
        hh[2] = (__bf16)f.z; hh[3] = (__bf16)f.w;
        *(bf16x4*)(su + i * EMB_DIM + lane * 4) = hh;
    }

    // ---- Build A fragments once (K=128 -> 4 chunks of 32), reused by all 11 matmuls ----
    // 16-bit A 16x32 layout: lane l<16: K[8h..8h+8) then K[16+8h..), h = l>>4
    const int m   = lane & 15;
    const int h16 = lane >> 4;
    FragBF afrag[4];
    #pragma unroll
    for (int c = 0; c < 4; ++c) {
        const int base = m * EMB_DIM + c * 32 + h16 * 8;
        afrag[c].h[0] = *(const bf16x8*)(su + base);       // K = 8h .. 8h+7
        afrag[c].h[1] = *(const bf16x8*)(su + base + 16);  // K = 16+8h .. 16+8h+7
    }

    const bool active = (lane < 8) || (lane >= 24);
    const int  reg    = (lane < 8) ? lane : lane - 24;

    float loss = 0.0f;

    // ---- mm==0: positive pair; mm=1..10: negatives ----
    #pragma unroll 1
    for (int mm = 0; mm < 1 + NEG_K; ++mm) {
        // Stage B tile row-major: sb[n*EMB_DIM + k] = row_n[k] (bf16)
        #pragma unroll
        for (int i = 0; i < TILE_M; ++i) {
            const int idx = (mm == 0)
                ? pos_v[b0 + i]
                : neg_v[(size_t)(b0 + i) * NEG_K + (mm - 1)];
            const float4 f = *((const float4*)(v_emb + (size_t)idx * EMB_DIM) + lane);
            bf16x4 hh;
            hh[0] = (__bf16)f.x; hh[1] = (__bf16)f.y;
            hh[2] = (__bf16)f.z; hh[3] = (__bf16)f.w;
            *(bf16x4*)(sb + i * EMB_DIM + lane * 4) = hh;
        }

        // Prefetch next iteration's 16 rows while this tile is consumed
        if (mm < NEG_K) {
            #pragma unroll
            for (int i = 0; i < TILE_M; ++i) {
                const int idx = neg_v[(size_t)(b0 + i) * NEG_K + mm];
                prefetch_row(v_emb + (size_t)idx * EMB_DIM, lane);
            }
        }

        v8f acc = {};
        #pragma unroll
        for (int c = 0; c < 4; ++c) {
            // B 32x16 layout: lane l holds column N=l&15, K[8h..8h+8) then K[16+8h..), h = l>>4
            FragBF bfrag;
            const __bf16* row = sb + m * EMB_DIM + c * 32 + h16 * 8;
            bfrag.h[0] = *(const bf16x8*)(row);
            bfrag.h[1] = *(const bf16x8*)(row + 16);
            acc = __builtin_amdgcn_wmma_f32_16x16x32_bf16(
                false, afrag[c].v, false, bfrag.v, (short)0, acc, false, false);
        }

        float d = diag_select(acc, reg);
        d = fminf(fmaxf(d, -10.0f), 10.0f);
        // pos: -log_sigmoid(x) = softplus(-x); neg: -log_sigmoid(-x) = softplus(x)
        const float term = (mm == 0) ? softplus_f(-d) : softplus_f(d);
        if (active) loss += term;
    }

    // ---- wave reduction over the 16 diagonal-holding lanes ----
    if (!active) loss = 0.0f;
    #pragma unroll
    for (int off = 16; off > 0; off >>= 1)
        loss += __shfl_xor(loss, off, 32);

    if (lane == 0) s_wavesum[w] = loss;
    __syncthreads();
    if (threadIdx.x == 0) {
        float s = 0.0f;
        #pragma unroll
        for (int j = 0; j < WAVES_PER_BLOCK; ++j) s += s_wavesum[j];
        block_sums[blockIdx.x] = s;   // no atomics: bit-deterministic
    }
}

__global__ void __launch_bounds__(256)
final_reduce_kernel(const float* __restrict__ block_sums, float* __restrict__ out, int n)
{
    __shared__ float buf[256];
    float s = 0.0f;
    for (int i = threadIdx.x; i < n; i += 256) s += block_sums[i];
    buf[threadIdx.x] = s;
    __syncthreads();
    #pragma unroll
    for (int off = 128; off > 0; off >>= 1) {
        if (threadIdx.x < off) buf[threadIdx.x] += buf[threadIdx.x + off];
        __syncthreads();
    }
    if (threadIdx.x == 0) out[0] = buf[0] * (1.0f / (float)BATCH);
}

extern "C" void kernel_launch(void* const* d_in, const int* in_sizes, int n_in,
                              void* d_out, int out_size, void* d_ws, size_t ws_size,
                              hipStream_t stream) {
    const float* u_emb = (const float*)d_in[0];
    const float* v_emb = (const float*)d_in[1];
    const int*   pos_u = (const int*)d_in[2];
    const int*   pos_v = (const int*)d_in[3];
    const int*   neg_v = (const int*)d_in[4];
    float* out = (float*)d_out;
    float* ws  = (float*)d_ws;     // 512 floats of scratch

    const int tiles  = BATCH / TILE_M;             // 4096 tiles of 16 batch elems
    const int blocks = tiles / WAVES_PER_BLOCK;    // 512 blocks x 256 threads

    sgns_loss_kernel<<<blocks, 32 * WAVES_PER_BLOCK, 0, stream>>>(
        u_emb, v_emb, pos_u, pos_v, neg_v, ws);
    final_reduce_kernel<<<1, 256, 0, stream>>>(ws, out, blocks);
}
